// RWSEEncoder_75230647157223
// MI455X (gfx1250) — matrix operand order repaired and checked by reading, not compiled, not run on hardware
//
#include <hip/hip_runtime.h>
#include <hip/hip_bf16.h>
#include <stdint.h>

// Problem constants (match reference)
#define TN    4096
#define WL    16
#define OD    8
#define EDGES 65536

// GEMM tiling
#define BM 128
#define BN 128
#define BK 32
#define LDT 40   // padded LDS row stride in elements (80B, 16B-aligned, conflict-free)

typedef __attribute__((ext_vector_type(8)))  __bf16 v8bf;
typedef __attribute__((ext_vector_type(16))) __bf16 v16bf;
typedef __attribute__((ext_vector_type(8)))  float  v8f;

__device__ __forceinline__ v16bf ld_frag16(const __bf16* p0, const __bf16* p1) {
  v8bf a = *(const v8bf*)p0;
  v8bf b = *(const v8bf*)p1;
  return __builtin_shufflevector(a, b, 0,1,2,3,4,5,6,7,8,9,10,11,12,13,14,15);
}

// ---------------- setup kernels ----------------

__global__ void init_kernel(float* __restrict__ deg, float* __restrict__ Af32,
                            __bf16* __restrict__ Xh0, __bf16* __restrict__ Xl0) {
  size_t i = (size_t)blockIdx.x * blockDim.x + threadIdx.x;
  if (i >= (size_t)TN * TN) return;
  Af32[i] = 0.0f;
  int r = (int)(i >> 12);
  int c = (int)(i & (TN - 1));
  Xh0[i] = (r == c) ? (__bf16)1.0f : (__bf16)0.0f;
  Xl0[i] = (__bf16)0.0f;
  if (i < TN) deg[i] = 0.0f;
}

__global__ void deg_kernel(const int* __restrict__ ei, float* __restrict__ deg) {
  int e = blockIdx.x * blockDim.x + threadIdx.x;
  if (e < EDGES) atomicAdd(&deg[ei[e]], 1.0f);
}

__global__ void invdeg_kernel(float* __restrict__ deg) {
  int i = blockIdx.x * blockDim.x + threadIdx.x;
  if (i < TN) deg[i] = 1.0f / fmaxf(deg[i], 1.0f);
}

// Pt[c][r] += 1/deg[r]  (duplicate edges coalesce via atomic add, like the reference)
__global__ void scatter_kernel(const int* __restrict__ ei, const float* __restrict__ invdeg,
                               float* __restrict__ Af32) {
  int e = blockIdx.x * blockDim.x + threadIdx.x;
  if (e >= EDGES) return;
  int r = ei[e];
  int c = ei[EDGES + e];
  atomicAdd(&Af32[(size_t)c * TN + r], invdeg[r]);
}

// Split f32 A into (hi, lo) bf16 pair; X(step1) = A; extract diag of step 1.
__global__ void split_kernel(const float* __restrict__ Af32,
                             __bf16* __restrict__ Ah, __bf16* __restrict__ Al,
                             __bf16* __restrict__ Xh, __bf16* __restrict__ Xl,
                             float* __restrict__ diag0) {
  size_t i = (size_t)blockIdx.x * blockDim.x + threadIdx.x;
  if (i >= (size_t)TN * TN) return;
  float a = Af32[i];
  __bf16 h = (__bf16)a;
  __bf16 l = (__bf16)(a - (float)h);
  Ah[i] = h; Al[i] = l;
  Xh[i] = h; Xl[i] = l;
  int r = (int)(i >> 12);
  int c = (int)(i & (TN - 1));
  if (r == c) diag0[r] = a;
}

// ---------------- split-precision bf16 WMMA GEMM ----------------
// Y(f32 accum) = A * X  with A,X given as bf16 (hi,lo) pairs:
//   Y ≈ Ah*Xh + Ah*Xl + Al*Xh        (error ~2^-16, f32-class over 16 steps)
// Output re-split into bf16 (hi,lo) for the next power step; diagonal captured.
__global__ __launch_bounds__(256)
void gemm_kernel(const __bf16* __restrict__ Ah, const __bf16* __restrict__ Al,
                 const __bf16* __restrict__ Xh, const __bf16* __restrict__ Xl,
                 __bf16* __restrict__ Yh, __bf16* __restrict__ Yl,
                 float* __restrict__ diag) {
  __shared__ __bf16 sAh[BM * LDT];
  __shared__ __bf16 sAl[BM * LDT];
  __shared__ __bf16 sXh[BN * LDT];   // transposed: [n][k]
  __shared__ __bf16 sXl[BN * LDT];

  const int tid  = threadIdx.x;
  const int lane = tid & 31;
  const int wave = tid >> 5;        // 8 waves
  const int wr   = wave >> 2;       // 0..1  (64-row slab)
  const int wc   = wave & 3;        // 0..3  (32-col slab)
  const int m_blk = blockIdx.y * BM;
  const int n_blk = blockIdx.x * BN;
  const int l16 = lane & 15;
  const int lh  = lane >> 4;

  v8f acc[4][2] = {};

  // A-tile load map: 128 rows x 32 k; thread -> (row, 16-elem half)
  const int ar = tid >> 1;
  const int ac = (tid & 1) << 4;
  // X-tile load map: 32 k x 128 n; thread -> (k, 8 cols), stored transposed
  const int xk = tid >> 4;
  const int xn = (tid & 15) << 3;

  for (int k0 = 0; k0 < TN; k0 += BK) {
    __syncthreads();
    {
      size_t g = (size_t)(m_blk + ar) * TN + (size_t)(k0 + ac);
      *(v8bf*)&sAh[ar * LDT + ac]     = *(const v8bf*)&Ah[g];
      *(v8bf*)&sAh[ar * LDT + ac + 8] = *(const v8bf*)&Ah[g + 8];
      *(v8bf*)&sAl[ar * LDT + ac]     = *(const v8bf*)&Al[g];
      *(v8bf*)&sAl[ar * LDT + ac + 8] = *(const v8bf*)&Al[g + 8];
      if (k0 + BK < TN) {            // hint next K panel (global_prefetch_b8)
        __builtin_prefetch(&Ah[g + BK], 0, 0);
        __builtin_prefetch(&Al[g + BK], 0, 0);
      }
    }
    {
      size_t g0 = (size_t)(k0 + xk) * TN + (size_t)(n_blk + xn);
      size_t g1 = (size_t)(k0 + xk + 16) * TN + (size_t)(n_blk + xn);
      v8bf h0 = *(const v8bf*)&Xh[g0];
      v8bf h1 = *(const v8bf*)&Xh[g1];
      v8bf l0 = *(const v8bf*)&Xl[g0];
      v8bf l1 = *(const v8bf*)&Xl[g1];
#pragma unroll
      for (int i = 0; i < 8; ++i) {
        sXh[(xn + i) * LDT + xk]      = h0[i];
        sXh[(xn + i) * LDT + xk + 16] = h1[i];
        sXl[(xn + i) * LDT + xk]      = l0[i];
        sXl[(xn + i) * LDT + xk + 16] = l1[i];
      }
      if (k0 + BK < TN) {
        __builtin_prefetch(&Xh[g0 + (size_t)BK * TN], 0, 0);
        __builtin_prefetch(&Xl[g0 + (size_t)BK * TN], 0, 0);
      }
    }
    __syncthreads();

    // B fragments (column-per-lane; X stored transposed so loads are contiguous)
    v16bf bh[2], bl[2];
#pragma unroll
    for (int ni = 0; ni < 2; ++ni) {
      const __bf16* pb = &sXh[(wc * 32 + ni * 16 + l16) * LDT + lh * 8];
      bh[ni] = ld_frag16(pb, pb + 16);
      const __bf16* pl = &sXl[(wc * 32 + ni * 16 + l16) * LDT + lh * 8];
      bl[ni] = ld_frag16(pl, pl + 16);
    }
#pragma unroll
    for (int mi = 0; mi < 4; ++mi) {
      const __bf16* pa  = &sAh[(wr * 64 + mi * 16 + l16) * LDT + lh * 8];
      v16bf fah = ld_frag16(pa, pa + 16);
      const __bf16* pal = &sAl[(wr * 64 + mi * 16 + l16) * LDT + lh * 8];
      v16bf fal = ld_frag16(pal, pal + 16);
#pragma unroll
      for (int ni = 0; ni < 2; ++ni) {
        acc[mi][ni] = __builtin_amdgcn_wmma_f32_16x16x32_bf16(
            false, fah, false, bh[ni], (short)0, acc[mi][ni], false, false);
        acc[mi][ni] = __builtin_amdgcn_wmma_f32_16x16x32_bf16(
            false, fah, false, bl[ni], (short)0, acc[mi][ni], false, false);
        acc[mi][ni] = __builtin_amdgcn_wmma_f32_16x16x32_bf16(
            false, fal, false, bh[ni], (short)0, acc[mi][ni], false, false);
      }
    }
  }

  // Epilogue: re-split f32 accumulators into bf16 (hi,lo); capture diagonal.
  // C/D layout: lane 0-15 -> N=lane, M=r; lane 16-31 -> N=lane-16, M=r+8.
#pragma unroll
  for (int mi = 0; mi < 4; ++mi) {
#pragma unroll
    for (int ni = 0; ni < 2; ++ni) {
      const int col = n_blk + wc * 32 + ni * 16 + l16;
#pragma unroll
      for (int r = 0; r < 8; ++r) {
        const int row = m_blk + wr * 64 + mi * 16 + r + (lh << 3);
        float y = acc[mi][ni][r];
        __bf16 h = (__bf16)y;
        size_t g = (size_t)row * TN + col;
        Yh[g] = h;
        Yl[g] = (__bf16)(y - (float)h);
        if (row == col) diag[row] = y;
      }
    }
  }
}

// ---------------- head: out = rw_diag @ W^T + b ----------------
__global__ void head_kernel(const float* __restrict__ diag, const float* __restrict__ W,
                            const float* __restrict__ b, float* __restrict__ out) {
  int i = blockIdx.x * blockDim.x + threadIdx.x;
  if (i >= TN) return;
  float a[OD];
#pragma unroll
  for (int d = 0; d < OD; ++d) a[d] = b[d];
#pragma unroll
  for (int k = 0; k < WL; ++k) {
    float v = diag[(size_t)k * TN + i];
#pragma unroll
    for (int d = 0; d < OD; ++d) a[d] += v * W[d * WL + k];
  }
#pragma unroll
  for (int d = 0; d < OD; ++d) out[i * OD + d] = a[d];
}

// ---------------- launch ----------------
extern "C" void kernel_launch(void* const* d_in, const int* in_sizes, int n_in,
                              void* d_out, int out_size, void* d_ws, size_t ws_size,
                              hipStream_t stream) {
  (void)in_sizes; (void)n_in; (void)out_size; (void)ws_size;
  const int*   ei = (const int*)d_in[0];     // edge_index [2,E]
  const float* W  = (const float*)d_in[2];   // [OD, WL]
  const float* bb = (const float*)d_in[3];   // [OD]
  float* out = (float*)d_out;

  // Workspace layout (~192.3 MB total):
  //   deg (16KB) | diag (256KB) | Ah | Al | Xh0 | Xl0 | Xh1 | Xl1  (6 x 32MB)
  //   Af32 (64MB f32) aliases [Xh1|Xl1] — consumed by split before GEMM writes there.
  char* ws = (char*)d_ws;
  float* deg  = (float*)ws;  ws += 16 * 1024;
  float* diag = (float*)ws;  ws += (size_t)WL * TN * sizeof(float);
  const size_t MB = (size_t)TN * TN * sizeof(__bf16);
  __bf16* Ah  = (__bf16*)ws; ws += MB;
  __bf16* Al  = (__bf16*)ws; ws += MB;
  __bf16* Xh0 = (__bf16*)ws; ws += MB;
  __bf16* Xl0 = (__bf16*)ws; ws += MB;
  __bf16* Xh1 = (__bf16*)ws; ws += MB;
  __bf16* Xl1 = (__bf16*)ws; ws += MB;
  float* Af32 = (float*)Xh1;

  const size_t NN = (size_t)TN * TN;
  const unsigned nb = (unsigned)((NN + 255) / 256);

  init_kernel<<<nb, 256, 0, stream>>>(deg, Af32, Xh0, Xl0);
  deg_kernel<<<EDGES / 256, 256, 0, stream>>>(ei, deg);
  invdeg_kernel<<<TN / 256, 256, 0, stream>>>(deg);
  scatter_kernel<<<EDGES / 256, 256, 0, stream>>>(ei, deg, Af32);
  split_kernel<<<nb, 256, 0, stream>>>(Af32, Ah, Al, Xh0, Xl0, diag);  // step 1 = A itself

  __bf16* xs[2][2] = {{Xh0, Xl0}, {Xh1, Xl1}};
  dim3 grid(TN / BN, TN / BM);
  int s = 0;
  for (int k = 1; k < WL; ++k) {   // 15 power steps via WMMA GEMM
    gemm_kernel<<<grid, 256, 0, stream>>>(Ah, Al, xs[s][0], xs[s][1],
                                          xs[1 - s][0], xs[1 - s][1],
                                          diag + (size_t)k * TN);
    s ^= 1;
  }
  head_kernel<<<TN / 256, 256, 0, stream>>>(diag, W, bb, out);
}
